// DSVTBlock_20091857011187
// MI455X (gfx1250) — matrix operand order repaired and checked by reading, not compiled, not run on hardware
//
#include <hip/hip_runtime.h>

typedef _Float16 h16;
typedef __attribute__((ext_vector_type(16))) _Float16 v16h;
typedef __attribute__((ext_vector_type(8)))  _Float16 v8h;
typedef __attribute__((ext_vector_type(8)))  float    v8f;
typedef int v4i __attribute__((vector_size(16)));   // matches builtin's param type

#define NVOX  147456
#define CDIM  192
#define NHEAD 8
#define DH    24
#define DFFN  768
#define SETSZ 36
#define NSETS 4096

#define BM 128
#define BN 64
#define BK 32

// ---- gfx1250 async global->LDS path, guarded so the file compiles either way
#if defined(__HIP_DEVICE_COMPILE__) && __has_builtin(__builtin_amdgcn_global_load_async_to_lds_b128)
#define HAVE_ASYNC_LDS 1
#else
#define HAVE_ASYNC_LDS 0
#endif

#if HAVE_ASYNC_LDS
__device__ __forceinline__ void async_cp16(const h16* g, h16* l) {
  // param0: global (AS1) v4i*  — global addresses are identity-mapped in flat
  // param1: LDS    (AS3) v4i*  — explicit addrspace cast from generic
  __builtin_amdgcn_global_load_async_to_lds_b128(
      (__attribute__((address_space(1))) v4i*)(unsigned long long)(const void*)g,
      (__attribute__((address_space(3))) v4i*)l, 0, 0);
}
__device__ __forceinline__ void async_wait0() {
#if __has_builtin(__builtin_amdgcn_s_wait_asynccnt)
  __builtin_amdgcn_s_wait_asynccnt(0);
#else
  asm volatile("s_wait_asynccnt 0" ::: "memory");
#endif
}
#endif

// ---------------------------------------------------------------------------
// fp16 WMMA GEMM:  D[M,Nout] = A[M,K] @ Wt[K,Nout] + bias   (Wt is K-major)
// 256 threads = 8 waves (4x2); block tile 128x64; each wave 2x2 WMMA tiles.
// ---------------------------------------------------------------------------
template<bool RELU, bool OUT16>
__global__ __launch_bounds__(256) void gemm_f16_wmma(
    const h16* __restrict__ A, const h16* __restrict__ Wt,
    const float* __restrict__ bias,
    h16* __restrict__ D16, float* __restrict__ D32,
    int M, int Nout, int K)
{
  __shared__ h16 lA[BM][BK];   // 128x32  [m][k]   8 KB
  __shared__ h16 lB[BK][BN];   // 32x64   [k][n]   4 KB

  const int bm   = blockIdx.y * BM;
  const int bn   = blockIdx.x * BN;
  const int tid  = threadIdx.x;
  const int lane = tid & 31;
  const int wmq  = (tid >> 5) & 3;   // wave M group (0..3)
  const int wnq  = (tid >> 5) >> 2;  // wave N group (0..1)

  // 16B staging chunks: A = 4096 halfs -> 2 chunks/thread, B = 2048 -> 1/thread
  const int ar0 = (2 * tid) >> 2,     ac0 = ((2 * tid) & 3) * 8;
  const int ar1 = (2 * tid + 1) >> 2, ac1 = ((2 * tid + 1) & 3) * 8;
  const int bk  = tid >> 3,           bc  = (tid & 7) * 8;

  v8f acc[2][2];
#pragma unroll
  for (int mi = 0; mi < 2; ++mi)
#pragma unroll
    for (int ni = 0; ni < 2; ++ni)
#pragma unroll
      for (int r = 0; r < 8; ++r) acc[mi][ni][r] = 0.f;

  const int ah = lane >> 4;       // K-half select for A fragment
  const int am = lane & 15;       // M row within tile

  for (int k0 = 0; k0 < K; k0 += BK) {
    const h16* gA0 = A  + (size_t)(bm + ar0) * K + (k0 + ac0);
    const h16* gA1 = A  + (size_t)(bm + ar1) * K + (k0 + ac1);
    const h16* gB  = Wt + (size_t)(k0 + bk) * Nout + (bn + bc);
#if HAVE_ASYNC_LDS
    async_cp16(gA0, &lA[ar0][ac0]);
    async_cp16(gA1, &lA[ar1][ac1]);
    async_cp16(gB,  &lB[bk][bc]);
    if (k0 + BK < K) {
      __builtin_prefetch(gA0 + BK, 0, 1);
      __builtin_prefetch(gB + (size_t)BK * Nout, 0, 1);
    }
    async_wait0();
#else
    const uint4 va0 = *(const uint4*)gA0;    // 3 loads in flight before any wait
    const uint4 va1 = *(const uint4*)gA1;
    const uint4 vb  = *(const uint4*)gB;
    if (k0 + BK < K) {
      __builtin_prefetch(gA0 + BK, 0, 1);
      __builtin_prefetch(gB + (size_t)BK * Nout, 0, 1);
    }
    *(uint4*)&lA[ar0][ac0] = va0;
    *(uint4*)&lA[ar1][ac1] = va1;
    *(uint4*)&lB[bk][bc]   = vb;
#endif
    __syncthreads();

    // A fragments: two contiguous 16B runs per lane (ISA 7.12.2 16-bit A layout)
    v16h af[2];
#pragma unroll
    for (int mi = 0; mi < 2; ++mi) {
      const int row = (wmq * 2 + mi) * 16 + am;
      const v8h a0 = *(const v8h*)&lA[row][ah * 8];
      const v8h a1 = *(const v8h*)&lA[row][16 + ah * 8];
      af[mi] = __builtin_shufflevector(a0, a1, 0, 1, 2, 3, 4, 5, 6, 7,
                                               8, 9, 10, 11, 12, 13, 14, 15);
    }
    // B fragments: lane&15 = N column, regs = K in same order as A (8h+s, 16+8h+s)
    v16h bf[2];
#pragma unroll
    for (int ni = 0; ni < 2; ++ni) {
      const int nb = (wnq * 2 + ni) * 16;
      v16h bt;
#pragma unroll
      for (int s = 0; s < 8; ++s) {
        bt[s]     = lB[ah * 8 + s][nb + am];
        bt[8 + s] = lB[16 + ah * 8 + s][nb + am];
      }
      bf[ni] = bt;
    }
#pragma unroll
    for (int mi = 0; mi < 2; ++mi)
#pragma unroll
      for (int ni = 0; ni < 2; ++ni)
        acc[mi][ni] = __builtin_amdgcn_wmma_f32_16x16x32_f16(
            false, af[mi], false, bf[ni], (short)0, acc[mi][ni], false, false);
    __syncthreads();
  }

  // D layout: VGPR r -> M = 8*(lane>>4)+r ; lane&15 -> N
#pragma unroll
  for (int mi = 0; mi < 2; ++mi) {
#pragma unroll
    for (int ni = 0; ni < 2; ++ni) {
      const int col = bn + (wnq * 2 + ni) * 16 + (lane & 15);
      const float bcol = bias[col];
#pragma unroll
      for (int r = 0; r < 8; ++r) {
        const int row = bm + (wmq * 2 + mi) * 16 + 8 * (lane >> 4) + r;
        float v = acc[mi][ni][r] + bcol;
        if (RELU) v = v > 0.f ? v : 0.f;
        if (OUT16) D16[(size_t)row * Nout + col] = (h16)v;
        else       D32[(size_t)row * Nout + col] = v;
      }
    }
  }
}

// ---------------------------------------------------------------------------
__global__ __launch_bounds__(256) void convert_transpose_f16(
    const float* __restrict__ in /*Nout x K*/, h16* __restrict__ out /*K x Nout*/,
    int Nout, int K)
{
  int i = blockIdx.x * blockDim.x + threadIdx.x;
  if (i >= Nout * K) return;
  const int n = i / K, k = i % K;
  out[(size_t)k * Nout + n] = (h16)in[i];
}

__global__ __launch_bounds__(256) void copy_f32(
    const float* __restrict__ in, float* __restrict__ out, size_t n)
{
  size_t i = (size_t)blockIdx.x * blockDim.x + threadIdx.x;
  if (i < n) out[i] = in[i];
}

// gather voxels into set order; emit f16 feat and f16 (feat+pos)
__global__ __launch_bounds__(256) void gather_kernel(
    const float* __restrict__ outbuf, const float* __restrict__ pos,
    const int* __restrict__ inds, const int* __restrict__ block_id, int set_id,
    h16* __restrict__ sf16, h16* __restrict__ qin16)
{
  size_t idx = (size_t)blockIdx.x * blockDim.x + threadIdx.x;
  if (idx >= (size_t)NVOX * CDIM) return;
  const int r = (int)(idx / CDIM), c = (int)(idx % CDIM);
  const int shift = block_id[0] & 1;
  const int vox = inds[(size_t)(shift * 2 + set_id) * NVOX + r];
  const float f = outbuf[(size_t)vox * CDIM + c];
  const float p = pos[(size_t)vox * CDIM + c];
  sf16[idx]  = (h16)f;
  qin16[idx] = (h16)(f + p);
}

// per-thread (set, head, query-row) attention; fp32 math on f16 q/k/v
__global__ __launch_bounds__(256) void attn_kernel(
    const h16* __restrict__ q, const h16* __restrict__ k, const h16* __restrict__ v,
    const unsigned char* __restrict__ masks, const int* __restrict__ block_id,
    int set_id, h16* __restrict__ o16)
{
  const int gid = blockIdx.x * blockDim.x + threadIdx.x;
  if (gid >= NSETS * NHEAD * SETSZ) return;
  const int qr   = gid % SETSZ;
  const int head = (gid / SETSZ) % NHEAD;
  const int set  = gid / (SETSZ * NHEAD);
  const int shift = block_id[0] & 1;
  const unsigned char* m =
      masks + ((size_t)(shift * 2 + set_id) * NSETS + set) * SETSZ;
  const size_t base = (size_t)set * SETSZ * CDIM + head * DH;

  float qv[DH];
  const h16* qp = q + base + (size_t)qr * CDIM;
#pragma unroll
  for (int d = 0; d < DH; ++d) qv[d] = (float)qp[d];

  const float scale = 0.20412414523193154f;  // 1/sqrt(24)
  float s[SETSZ];
  float mx = -1e30f;
  for (int j = 0; j < SETSZ; ++j) {
    const h16* kp = k + base + (size_t)j * CDIM;
    float dot = 0.f;
#pragma unroll
    for (int d = 0; d < DH; ++d) dot += qv[d] * (float)kp[d];
    dot *= scale;
    if (m[j]) dot = -1e9f;
    s[j] = dot;
    mx = fmaxf(mx, dot);
  }
  float sum = 0.f;
  float accv[DH];
#pragma unroll
  for (int d = 0; d < DH; ++d) accv[d] = 0.f;
  for (int j = 0; j < SETSZ; ++j) {
    const float e = __expf(s[j] - mx);
    sum += e;
    const h16* vp = v + base + (size_t)j * CDIM;
#pragma unroll
    for (int d = 0; d < DH; ++d) accv[d] += e * (float)vp[d];
  }
  const float inv = 1.f / sum;
  h16* op = o16 + base + (size_t)qr * CDIM;
#pragma unroll
  for (int d = 0; d < DH; ++d) op[d] = (h16)(accv[d] * inv);
}

// permutation scatter + residual + LayerNorm1; one wave32 per set-order row
__global__ __launch_bounds__(256) void ln1_scatter_kernel(
    const float* __restrict__ src2_set, const float* __restrict__ outbuf,
    const int* __restrict__ inds, const int* __restrict__ block_id, int set_id,
    const float* __restrict__ w, const float* __restrict__ b,
    float* __restrict__ x, h16* __restrict__ x16)
{
  const int wid  = (blockIdx.x * blockDim.x + threadIdx.x) >> 5;
  const int lane = threadIdx.x & 31;
  if (wid >= NVOX) return;
  const int shift = block_id[0] & 1;
  const int vox = inds[(size_t)(shift * 2 + set_id) * NVOX + wid];

  float vals[6];
  float sum = 0.f;
#pragma unroll
  for (int j = 0; j < 6; ++j) {
    const int c = lane + 32 * j;
    const float t = outbuf[(size_t)vox * CDIM + c] + src2_set[(size_t)wid * CDIM + c];
    vals[j] = t; sum += t;
  }
#pragma unroll
  for (int off = 16; off; off >>= 1) sum += __shfl_xor(sum, off);
  const float mu = sum * (1.f / CDIM);
  float var = 0.f;
#pragma unroll
  for (int j = 0; j < 6; ++j) { const float d = vals[j] - mu; var += d * d; }
#pragma unroll
  for (int off = 16; off; off >>= 1) var += __shfl_xor(var, off);
  const float rs = rsqrtf(var * (1.f / CDIM) + 1e-5f);
#pragma unroll
  for (int j = 0; j < 6; ++j) {
    const int c = lane + 32 * j;
    const float y = (vals[j] - mu) * rs * w[c] + b[c];
    x[(size_t)vox * CDIM + c]   = y;
    x16[(size_t)vox * CDIM + c] = (h16)y;
  }
}

// fused LN2(x+ffn) then LN3(x2+identity); one wave32 per voxel row
__global__ __launch_bounds__(256) void ln23_kernel(
    const float* __restrict__ x, const float* __restrict__ ffn,
    const float* __restrict__ w2, const float* __restrict__ b2,
    const float* __restrict__ w3, const float* __restrict__ b3,
    float* __restrict__ outbuf)
{
  const int wid  = (blockIdx.x * blockDim.x + threadIdx.x) >> 5;
  const int lane = threadIdx.x & 31;
  if (wid >= NVOX) return;

  float vals[6];
  float sum = 0.f;
#pragma unroll
  for (int j = 0; j < 6; ++j) {
    const int c = lane + 32 * j;
    const float t = x[(size_t)wid * CDIM + c] + ffn[(size_t)wid * CDIM + c];
    vals[j] = t; sum += t;
  }
#pragma unroll
  for (int off = 16; off; off >>= 1) sum += __shfl_xor(sum, off);
  float mu = sum * (1.f / CDIM);
  float var = 0.f;
#pragma unroll
  for (int j = 0; j < 6; ++j) { const float d = vals[j] - mu; var += d * d; }
#pragma unroll
  for (int off = 16; off; off >>= 1) var += __shfl_xor(var, off);
  float rs = rsqrtf(var * (1.f / CDIM) + 1e-5f);

  sum = 0.f;
#pragma unroll
  for (int j = 0; j < 6; ++j) {
    const int c = lane + 32 * j;
    const float x2 = (vals[j] - mu) * rs * w2[c] + b2[c];
    const float t = x2 + outbuf[(size_t)wid * CDIM + c];
    vals[j] = t; sum += t;
  }
#pragma unroll
  for (int off = 16; off; off >>= 1) sum += __shfl_xor(sum, off);
  mu = sum * (1.f / CDIM);
  var = 0.f;
#pragma unroll
  for (int j = 0; j < 6; ++j) { const float d = vals[j] - mu; var += d * d; }
#pragma unroll
  for (int off = 16; off; off >>= 1) var += __shfl_xor(var, off);
  rs = rsqrtf(var * (1.f / CDIM) + 1e-5f);
#pragma unroll
  for (int j = 0; j < 6; ++j) {
    const int c = lane + 32 * j;
    outbuf[(size_t)wid * CDIM + c] = (vals[j] - mu) * rs * w3[c] + b3[c];
  }
}

// ---------------------------------------------------------------------------
extern "C" void kernel_launch(void* const* d_in, const int* in_sizes, int n_in,
                              void* d_out, int out_size, void* d_ws, size_t ws_size,
                              hipStream_t stream)
{
  (void)in_sizes; (void)n_in; (void)out_size; (void)ws_size;
  const float*         src        = (const float*)d_in[0];
  const int*           inds       = (const int*)d_in[1];
  const unsigned char* masks      = (const unsigned char*)d_in[2];
  const float*         pos        = (const float*)d_in[3];
  const int*           block_id   = (const int*)d_in[4];
  const float*         in_proj_w  = (const float*)d_in[5];
  const float*         in_proj_b  = (const float*)d_in[6];
  const float*         out_proj_w = (const float*)d_in[7];
  const float*         out_proj_b = (const float*)d_in[8];
  const float*         lin1_w     = (const float*)d_in[9];
  const float*         lin1_b     = (const float*)d_in[10];
  const float*         lin2_w     = (const float*)d_in[11];
  const float*         lin2_b     = (const float*)d_in[12];
  const float*         n1w = (const float*)d_in[13];
  const float*         n1b = (const float*)d_in[14];
  const float*         n2w = (const float*)d_in[15];
  const float*         n2b = (const float*)d_in[16];
  const float*         n3w = (const float*)d_in[17];
  const float*         n3b = (const float*)d_in[18];

  const size_t NC = (size_t)NVOX * CDIM;
  char* p = (char*)d_ws;
  auto alloc = [&](size_t bytes) -> char* {
    char* r = p; p += (bytes + 255) & ~(size_t)255; return r;
  };
  float* buf_out  = (float*)alloc(NC * 4);
  float* buf_src2 = (float*)alloc(NC * 4);
  float* buf_x    = (float*)alloc(NC * 4);
  float* buf_ffn  = (float*)alloc(NC * 4);
  h16* qin16 = (h16*)alloc(NC * 2);
  h16* sf16  = (h16*)alloc(NC * 2);
  h16* q16   = (h16*)alloc(NC * 2);
  h16* k16   = (h16*)alloc(NC * 2);
  h16* v16   = (h16*)alloc(NC * 2);
  h16* o16   = (h16*)alloc(NC * 2);
  h16* x16   = (h16*)alloc(NC * 2);
  h16* hid16 = (h16*)alloc((size_t)NVOX * DFFN * 2);
  h16* wt_in  = (h16*)alloc((size_t)2 * 3 * CDIM * CDIM * 2);  // K-major slices
  h16* wt_out = (h16*)alloc((size_t)2 * CDIM * CDIM * 2);
  h16* wt_l1  = (h16*)alloc((size_t)2 * DFFN * CDIM * 2);
  h16* wt_l2  = (h16*)alloc((size_t)2 * CDIM * DFFN * 2);

  const int CC = CDIM * CDIM, DC = DFFN * CDIM;
  const int bCC = (CC + 255) / 256, bDC = (DC + 255) / 256;
  for (int i = 0; i < 2; ++i) {   // per-slice transposed f16 weights (static)
    convert_transpose_f16<<<bCC, 256, 0, stream>>>(in_proj_w + (size_t)i*3*CC,        wt_in  + (size_t)i*3*CC,        CDIM, CDIM);
    convert_transpose_f16<<<bCC, 256, 0, stream>>>(in_proj_w + (size_t)i*3*CC + CC,   wt_in  + (size_t)i*3*CC + CC,   CDIM, CDIM);
    convert_transpose_f16<<<bCC, 256, 0, stream>>>(in_proj_w + (size_t)i*3*CC + 2*CC, wt_in  + (size_t)i*3*CC + 2*CC, CDIM, CDIM);
    convert_transpose_f16<<<bCC, 256, 0, stream>>>(out_proj_w + (size_t)i*CC,         wt_out + (size_t)i*CC,          CDIM, CDIM);
    convert_transpose_f16<<<bDC, 256, 0, stream>>>(lin1_w + (size_t)i*DC,             wt_l1  + (size_t)i*DC,          DFFN, CDIM);
    convert_transpose_f16<<<bDC, 256, 0, stream>>>(lin2_w + (size_t)i*DC,             wt_l2  + (size_t)i*DC,          CDIM, DFFN);
  }
  copy_f32<<<(int)((NC + 255) / 256), 256, 0, stream>>>(src, buf_out, NC);

  const dim3 gemm_blk(256);
  const dim3 gC(CDIM / BN, NVOX / BM);   // (3, 1152)
  const dim3 gF(DFFN / BN, NVOX / BM);   // (12, 1152)
  const int elem_blocks = (int)((NC + 255) / 256);
  const int attn_blocks = (NSETS * NHEAD * SETSZ + 255) / 256;
  const int wave_blocks = (NVOX * 32 + 255) / 256;

  for (int i = 0; i < 2; ++i) {
    gather_kernel<<<elem_blocks, 256, 0, stream>>>(
        buf_out, pos + (size_t)i * NC, inds, block_id, i, sf16, qin16);

    const h16* wq = wt_in + (size_t)i * 3 * CC;
    gemm_f16_wmma<false, true><<<gC, gemm_blk, 0, stream>>>(
        qin16, wq,          in_proj_b + (size_t)i * 3 * CDIM,
        q16, nullptr, NVOX, CDIM, CDIM);
    gemm_f16_wmma<false, true><<<gC, gemm_blk, 0, stream>>>(
        qin16, wq + CC,     in_proj_b + (size_t)i * 3 * CDIM + CDIM,
        k16, nullptr, NVOX, CDIM, CDIM);
    gemm_f16_wmma<false, true><<<gC, gemm_blk, 0, stream>>>(
        sf16,  wq + 2 * CC, in_proj_b + (size_t)i * 3 * CDIM + 2 * CDIM,
        v16, nullptr, NVOX, CDIM, CDIM);

    attn_kernel<<<attn_blocks, 256, 0, stream>>>(
        q16, k16, v16, masks, block_id, i, o16);

    gemm_f16_wmma<false, false><<<gC, gemm_blk, 0, stream>>>(
        o16, wt_out + (size_t)i * CC, out_proj_b + (size_t)i * CDIM,
        nullptr, buf_src2, NVOX, CDIM, CDIM);

    ln1_scatter_kernel<<<wave_blocks, 256, 0, stream>>>(
        buf_src2, buf_out, inds, block_id, i,
        n1w + (size_t)i * CDIM, n1b + (size_t)i * CDIM, buf_x, x16);

    gemm_f16_wmma<true, true><<<gF, gemm_blk, 0, stream>>>(
        x16, wt_l1 + (size_t)i * DC, lin1_b + (size_t)i * DFFN,
        hid16, nullptr, NVOX, DFFN, CDIM);
    gemm_f16_wmma<false, false><<<gC, gemm_blk, 0, stream>>>(
        hid16, wt_l2 + (size_t)i * DC, lin2_b + (size_t)i * CDIM,
        nullptr, buf_ffn, NVOX, CDIM, DFFN);

    ln23_kernel<<<wave_blocks, 256, 0, stream>>>(
        buf_x, buf_ffn,
        n2w + (size_t)i * CDIM, n2b + (size_t)i * CDIM,
        n3w + (size_t)i * CDIM, n3b + (size_t)i * CDIM, buf_out);
  }

  copy_f32<<<(int)((NC + 255) / 256), 256, 0, stream>>>(buf_out, (float*)d_out, NC);
}
